// Attention_54322746359846
// MI455X (gfx1250) — matrix operand order repaired and checked by reading, not compiled, not run on hardware
//
#include <hip/hip_runtime.h>

// ---------------------------------------------------------------------------
// Attention reference collapses algebraically:
//   attn = softmax(masked, axis=-1)  =>  each (t,n) row sums to 1
//   head_w = attn.sum(axis=(0,2))    =>  == N_NODES (4096) for every head
//   y = v * 4096;  out = y.reshape(N,512) @ w_proj + b
// => out = x @ (4096 * W_v @ w_proj) + b, where W_v[d, h*128+hd] = w_qkv[2,h,d,hd].
// adj (64 MB) and the 549-GFLOP adj@scores contraction are never needed.
// Remaining work: 512^3 pre-GEMM + 4096x512x512 GEMM, run via v_wmma f16.
// ---------------------------------------------------------------------------

typedef _Float16 f16;
typedef _Float16 v16h __attribute__((ext_vector_type(16)));
typedef _Float16 v8h  __attribute__((ext_vector_type(8)));
typedef float    v8f  __attribute__((ext_vector_type(8)));

#define N_NODES 4096
#define DIM     512

// ---- f32 -> f16 conversions ------------------------------------------------

__global__ void cvt_x_kernel(const float* __restrict__ x, f16* __restrict__ x16) {
    int i = blockIdx.x * blockDim.x + threadIdx.x;   // [0, 4096*512)
    x16[i] = (f16)x[i];
}

// W_v scaled by 4096 (the head_w factor), laid out row-major [d, c], c = h*128+hd
__global__ void cvt_wv_kernel(const float* __restrict__ wqkv, f16* __restrict__ wv16) {
    int i = blockIdx.x * blockDim.x + threadIdx.x;   // [0, 512*512)
    int d = i >> 9, c = i & 511;
    int h = c >> 7, hd = c & 127;
    const int vbase = 2 * 4 * DIM * 128;             // s=2 (V) slice of w_qkv
    float val = wqkv[vbase + (h * DIM + d) * 128 + hd];
    wv16[i] = (f16)(4096.0f * val);
}

// w_proj transposed: wpT[o, k] = w_proj[k, o] so WMMA B loads are K-contiguous
__global__ void cvt_wpT_kernel(const float* __restrict__ wp, f16* __restrict__ wpT) {
    int i = blockIdx.x * blockDim.x + threadIdx.x;   // [0, 512*512)
    int o = i >> 9, c = i & 511;
    wpT[i] = (f16)wp[c * DIM + o];                   // i == o*512 + c
}

// ---- WMMA core: one wave computes a 16(M) x 64(N) strip of C ---------------
// A : row-major [M, K] f16; Bt: row-major [N, K] f16 (i.e. B transposed).
// A-operand layout (ISA 7.12.2, 16-bit A 16x32): lane l -> row l&15,
//   halves 0..7 = K(8*(l>>4))..+7, halves 8..15 = K(16+8*(l>>4))..+7.
// B-operand layout (same K order as A): lane l -> col l&15,
//   halves 0..7 = K(8*(l>>4))..+7, halves 8..15 = K(16+8*(l>>4))..+7.
__device__ __forceinline__ void wmma_gemm_16x64(
    const f16* __restrict__ A, const f16* __restrict__ Bt,
    int K, int m0, int n0, v8f c[4]) {
    const int lane = threadIdx.x & 31;
    const int r    = lane & 15;
    const int hi   = lane >> 4;
    const f16* arow = A  + (m0 + r) * K + hi * 8;
    const f16* brow = Bt + (n0 + r) * K + hi * 8;
    for (int k = 0; k < K; k += 32) {
        v8h alo = *(const v8h*)(arow + k);
        v8h ahi = *(const v8h*)(arow + k + 16);
        v16h a = __builtin_shufflevector(alo, ahi,
                 0,1,2,3,4,5,6,7,8,9,10,11,12,13,14,15);
#pragma unroll
        for (int t = 0; t < 4; ++t) {
            const f16* bp = brow + t * 16 * K + k;
            v8h blo = *(const v8h*)(bp);
            v8h bhi = *(const v8h*)(bp + 16);
            v16h b = __builtin_shufflevector(blo, bhi,
                     0,1,2,3,4,5,6,7,8,9,10,11,12,13,14,15);
            // (neg_a, A, neg_b, B, c_mod, C, reuse_a, reuse_b)
            c[t] = __builtin_amdgcn_wmma_f32_16x16x32_f16(
                       false, a, false, b, (short)0, c[t], false, false);
        }
    }
}

// Stage A: WcombT[o, d] = (4096*W_v @ w_proj)^T  (stored transposed so it can
// feed stage B as a K-contiguous B operand), f16 output.
__global__ __launch_bounds__(32) void gemm_wcomb_kernel(
    const f16* __restrict__ Wv, const f16* __restrict__ WpT,
    f16* __restrict__ WcT) {
    const int m0 = blockIdx.x * 16;   // d
    const int n0 = blockIdx.y * 64;   // o
    v8f c[4];
#pragma unroll
    for (int t = 0; t < 4; ++t)
#pragma unroll
        for (int g = 0; g < 8; ++g) c[t][g] = 0.0f;

    wmma_gemm_16x64(Wv, WpT, DIM, m0, n0, c);

    // C layout: VGPR g -> (M = g + 8*(lane>>4), N = lane&15)
    const int lane = threadIdx.x & 31;
    const int n  = lane & 15;
    const int mb = (lane >> 4) * 8;
#pragma unroll
    for (int t = 0; t < 4; ++t) {
        v8h hv;
#pragma unroll
        for (int g = 0; g < 8; ++g) hv[g] = (f16)c[t][g];
        // store transposed: row o = n0+t*16+n, 8 contiguous d values
        *(v8h*)(WcT + (n0 + t * 16 + n) * DIM + m0 + mb) = hv;
    }
}

// Stage B: out[n, o] = x16 @ Wcomb + b_proj, f32 output.
__global__ __launch_bounds__(32) void gemm_out_kernel(
    const f16* __restrict__ X16, const f16* __restrict__ WcT,
    const float* __restrict__ bias, float* __restrict__ out) {
    const int m0 = blockIdx.x * 16;   // node row
    const int n0 = blockIdx.y * 64;   // output feature
    v8f c[4];
#pragma unroll
    for (int t = 0; t < 4; ++t)
#pragma unroll
        for (int g = 0; g < 8; ++g) c[t][g] = 0.0f;

    wmma_gemm_16x64(X16, WcT, DIM, m0, n0, c);

    const int lane = threadIdx.x & 31;
    const int n  = lane & 15;
    const int mb = (lane >> 4) * 8;
#pragma unroll
    for (int t = 0; t < 4; ++t) {
        const int o = n0 + t * 16 + n;
        const float bv = bias[o];
#pragma unroll
        for (int g = 0; g < 8; ++g)
            out[(m0 + mb + g) * DIM + o] = c[t][g] + bv;
    }
}

// ---------------------------------------------------------------------------

extern "C" void kernel_launch(void* const* d_in, const int* in_sizes, int n_in,
                              void* d_out, int out_size, void* d_ws, size_t ws_size,
                              hipStream_t stream) {
    (void)in_sizes; (void)n_in; (void)out_size; (void)ws_size;
    const float* x     = (const float*)d_in[0];
    // d_in[1] = adj : algebraically eliminated (softmax rows sum to 1)
    const float* wqkv  = (const float*)d_in[2];
    const float* wproj = (const float*)d_in[3];
    const float* bproj = (const float*)d_in[4];
    float* out = (float*)d_out;

    char* ws = (char*)d_ws;
    f16* x16  = (f16*)(ws);                        // 4096*512*2 = 4 MiB
    f16* wv16 = (f16*)(ws + 4194304);              // 512 KiB
    f16* wpT  = (f16*)(ws + 4194304 + 524288);     // 512 KiB
    f16* wcT  = (f16*)(ws + 4194304 + 1048576);    // 512 KiB

    cvt_x_kernel  <<<(N_NODES * DIM) / 256, 256, 0, stream>>>(x, x16);
    cvt_wv_kernel <<<(DIM * DIM) / 256,     256, 0, stream>>>(wqkv, wv16);
    cvt_wpT_kernel<<<(DIM * DIM) / 256,     256, 0, stream>>>(wproj, wpT);

    gemm_wcomb_kernel<<<dim3(DIM / 16, DIM / 64),     32, 0, stream>>>(wv16, wpT, wcT);
    gemm_out_kernel  <<<dim3(N_NODES / 16, DIM / 64), 32, 0, stream>>>(x16, wcT, bproj, out);
}
